// GVGG_12652973654224
// MI455X (gfx1250) — matrix operand order repaired and checked by reading, not compiled, not run on hardware
//
#include <hip/hip_runtime.h>
#include <math.h>

// ---------------------------------------------------------------------------
// MI455X (gfx1250) implementation of the 13-layer GCN/GAT + MLP reference.
//  - Dense GEMMs: bf16 A/B, f32 accumulate via v_wmma_f32_16x16x32_bf16.
//    Block = 8 waves sharing one 64-wide N block; B K-slices staged in LDS
//    (double-buffered, bank-conflict-free padded rows), each wave computes a
//    32x64 register tile (2x4 grid of 16x16 WMMA tiles) -> B is read from L2
//    once per block instead of once per wave (8x L2 traffic cut).
//  - All activations fit in the 192MB L2 (41MB per 20000x512 buffer), so the
//    random edge gather/scatter (fp32 atomics) is L2-resident.
// ---------------------------------------------------------------------------

typedef __attribute__((ext_vector_type(16))) __bf16 v16bf;
typedef __attribute__((ext_vector_type(8)))  float  v8f;
typedef __attribute__((ext_vector_type(4)))  float  v4f;

#define TPB 256
#define WPB 8                 // waves per block (share one N block)
#define LDS_ROWB 80           // padded LDS row stride in bytes (64B data + 16B)
#define BN_ROWS 128
#define LRELU_SLOPE 0.2f
#define BN_EPS 1e-5f

__device__ __forceinline__ void atomic_max_float(float* addr, float v) {
    // order-preserving int encoding: works for mixed-sign IEEE floats
    if (v >= 0.0f) atomicMax((int*)addr, __float_as_int(v));
    else           atomicMin((unsigned int*)addr, __float_as_uint(v));
}

// ------------------------------- utility ----------------------------------
__global__ void k_fill(float* p, float v, int n) {
    int i = blockIdx.x * blockDim.x + threadIdx.x;
    if (i < n) p[i] = v;
}

// F must be a power of two
__global__ void k_fill_bias(float* out, const float* __restrict__ bias, int rows, int F) {
    int i = blockIdx.x * blockDim.x + threadIdx.x;
    if (i < rows * F) out[i] = bias[i & (F - 1)];
}

__global__ void k_cvt_bf16(const float* __restrict__ a, __bf16* __restrict__ b, int n) {
    int i = blockIdx.x * blockDim.x + threadIdx.x;
    if (i < n) b[i] = (__bf16)a[i];
}

// W[k][n] (row-major K x Nf, fp32) -> WT[n][k] (row-major Nf x K, bf16)
__global__ void k_wt_bf16(const float* __restrict__ W, __bf16* __restrict__ WT, int K, int Nf) {
    int i = blockIdx.x * blockDim.x + threadIdx.x;
    if (i < K * Nf) {
        int k = i / Nf, n = i - k * Nf;
        WT[(size_t)n * K + k] = (__bf16)W[i];
    }
}

// ------------------------- degree / GCN norm ------------------------------
__global__ void k_deg(const int* __restrict__ dst, int E, float* deg) {
    int i = blockIdx.x * blockDim.x + threadIdx.x;
    if (i < E) atomicAdd(deg + dst[i], 1.0f);
}

__global__ void k_rsqrt_inplace(float* p, int n) {
    int i = blockIdx.x * blockDim.x + threadIdx.x;
    if (i < n) p[i] = rsqrtf(fmaxf(p[i], 1.0f));
}

// ------------------------- layer-0 GEMM (K=2) -----------------------------
__global__ void k_gemm_k2(const float* __restrict__ X, const float* __restrict__ W,
                          float* __restrict__ HW, int rows, int Nf) {
    int i = blockIdx.x * blockDim.x + threadIdx.x;
    if (i < rows * Nf) {
        int r = i / Nf, o = i - r * Nf;
        HW[i] = X[2 * r] * W[o] + X[2 * r + 1] * W[Nf + o];
    }
}

// ---------- WMMA bf16 GEMM: 8 waves/block share B via LDS staging ---------
// A: [M x K] row-major bf16. BT: [Nf x K] row-major bf16 (i.e., B transposed).
// C: [M x Nf] fp32 (+bias, optional ReLU).
// Requires K%32==0, M%32==0, Nf%64==0 (true for all WMMA-path shapes here).
// Fragment layouts per CDNA5 ISA 7.12.2:
//   A (16x32 bf16):  lane m=lane&15, half=lane>>4; elem j -> K = (j>>3)*16 + half*8 + (j&7)
//   B (32x16 bf16):  lane n=lane&15, half=lane>>4; elem j -> K = (j>>3)*16 + half*8 + (j&7)
//   C (16x16 f32):   lane n=lane&15, half=lane>>4; vgpr v -> M = v + half*8
__global__ __launch_bounds__(TPB)
void k_gemm_bf16(const __bf16* __restrict__ A, const __bf16* __restrict__ BT,
                 const float* __restrict__ bias, float* __restrict__ C,
                 int M, int K, int Nf, int relu) {
    // double-buffered B K-slice: 64 BT rows x 32 halfs, padded to 80B rows
    __shared__ alignas(16) char lds_b[2][64 * LDS_ROWB];

    int tid    = threadIdx.x;
    int waveid = tid >> 5;
    int lane   = tid & 31;

    int nblocks = Nf >> 6;                  // 64-wide N blocks
    int blk_m = blockIdx.x / nblocks;
    int tn    = blockIdx.x - blk_m * nblocks;

    int mtiles = M >> 5;                    // 32-row wave tiles
    int tm_raw = blk_m * WPB + waveid;
    int valid  = tm_raw < mtiles;
    int tm     = valid ? tm_raw : mtiles - 1;   // clamp: stay alive for barriers

    int m    = lane & 15;
    int half = lane >> 4;

    const __bf16* arow0 = A + (size_t)(tm * 32 + m) * K + half * 8;
    const __bf16* arow1 = arow0 + (size_t)16 * K;

    // staging: thread t loads BT[tn*64 + t/4][k0 + (t&3)*8 .. +7]  (16B)
    const __bf16* bsrc = BT + (size_t)(tn * 64 + (tid >> 2)) * K + (tid & 3) * 8;
    const int lrow = tid >> 2;
    const int lcol = (tid & 3) * 16;        // bytes

    v8f acc[2][4];
#pragma unroll
    for (int mi = 0; mi < 2; ++mi)
#pragma unroll
        for (int ni = 0; ni < 4; ++ni) acc[mi][ni] = (v8f){};

    const int nk = K >> 5;

    // preload K-slice 0 into buffer 0
    {
        v4f st = *(const v4f*)(bsrc);
        *(v4f*)&lds_b[0][lrow * LDS_ROWB + lcol] = st;
    }
    __syncthreads();

    for (int kb = 0; kb < nk; ++kb) {
        const int cur = kb & 1;
        const int k0  = kb * 32;

        v4f nxt;
        if (kb + 1 < nk)                                  // uniform branch
            nxt = *(const v4f*)(bsrc + (size_t)(kb + 1) * 32);

        if (k0 + 64 < K) {                                // WGP-scope prefetch of A stream
            __builtin_prefetch(arow0 + k0 + 64, 0, 3);    // global_prefetch_b8
            __builtin_prefetch(arow1 + k0 + 64, 0, 3);
        }

        union { v16bf v; v4f q[2]; } a0, a1, bfr[4];
        a0.q[0] = *(const v4f*)(arow0 + k0);
        a0.q[1] = *(const v4f*)(arow0 + k0 + 16);
        a1.q[0] = *(const v4f*)(arow1 + k0);
        a1.q[1] = *(const v4f*)(arow1 + k0 + 16);
#pragma unroll
        for (int ni = 0; ni < 4; ++ni) {
            // lane reads BT row (ni*16 + m), bytes [half*16, half*16+15] and [32+half*16, 32+half*16+15]
            bfr[ni].q[0] = *(const v4f*)&lds_b[cur][(ni * 16 + m) * LDS_ROWB + half * 16];
            bfr[ni].q[1] = *(const v4f*)&lds_b[cur][(ni * 16 + m) * LDS_ROWB + half * 16 + 32];
        }
#pragma unroll
        for (int ni = 0; ni < 4; ++ni) {
            acc[0][ni] = __builtin_amdgcn_wmma_f32_16x16x32_bf16(
                false, a0.v, false, bfr[ni].v, (short)0, acc[0][ni], false, false);
            acc[1][ni] = __builtin_amdgcn_wmma_f32_16x16x32_bf16(
                false, a1.v, false, bfr[ni].v, (short)0, acc[1][ni], false, false);
        }

        if (kb + 1 < nk) {
            *(v4f*)&lds_b[cur ^ 1][lrow * LDS_ROWB + lcol] = nxt;
            __syncthreads();
        }
    }

    if (valid) {
#pragma unroll
        for (int mi = 0; mi < 2; ++mi) {
#pragma unroll
            for (int ni = 0; ni < 4; ++ni) {
                int n = tn * 64 + ni * 16 + m;
                float bv = bias ? bias[n] : 0.0f;
                float* crow = C + (size_t)(tm * 32 + mi * 16 + half * 8) * Nf + n;
#pragma unroll
                for (int v = 0; v < 8; ++v) {
                    float r = acc[mi][ni][v] + bv;
                    if (relu) r = fmaxf(r, 0.0f);
                    crow[(size_t)v * Nf] = r;
                }
            }
        }
    }
}

// --------------------------- GCN aggregation ------------------------------
// out[dst] += dinv[src]*dinv[dst] * HW[src]   (wave per edge, lanes over F)
__global__ void k_gcn_agg(const int* __restrict__ src, const int* __restrict__ dst,
                          int E, int ET, const float* __restrict__ dinv,
                          const float* __restrict__ HW, float* __restrict__ out, int F) {
    int wid  = (blockIdx.x * blockDim.x + threadIdx.x) >> 5;
    int lane = threadIdx.x & 31;
    if (wid >= ET) return;
    int s = wid < E ? src[wid] : wid - E;
    int d = wid < E ? dst[wid] : wid - E;
    float nrm = dinv[s] * dinv[d];
    const float* hr = HW + (size_t)s * F;
    float* orow = out + (size_t)d * F;
    for (int f = lane; f < F; f += 32) atomicAdd(orow + f, nrm * hr[f]);
}

// ------------------------------ GAT kernels -------------------------------
// es[n]=dot(HW[n],a_s), ed[n]=dot(HW[n],a_d)   (wave per node)
__global__ void k_gat_dots(const float* __restrict__ HW, const float* __restrict__ as_,
                           const float* __restrict__ ad_, float* __restrict__ es,
                           float* __restrict__ ed, int rows, int F) {
    int wid  = (blockIdx.x * blockDim.x + threadIdx.x) >> 5;
    int lane = threadIdx.x & 31;
    if (wid >= rows) return;
    const float* row = HW + (size_t)wid * F;
    float a = 0.0f, b = 0.0f;
    for (int f = lane; f < F; f += 32) {
        float v = row[f];
        a += v * as_[f];
        b += v * ad_[f];
    }
#pragma unroll
    for (int mk = 16; mk > 0; mk >>= 1) {
        a += __shfl_xor(a, mk, 32);
        b += __shfl_xor(b, mk, 32);
    }
    if (lane == 0) { es[wid] = a; ed[wid] = b; }
}

// pass1: e = leaky_relu(es[src]+ed[dst]); segment max into mbuf[dst]
__global__ void k_gat_e1(const int* __restrict__ src, const int* __restrict__ dst,
                         int E, int ET, const float* __restrict__ es,
                         const float* __restrict__ ed, float* __restrict__ ebuf,
                         float* __restrict__ mbuf) {
    int i = blockIdx.x * blockDim.x + threadIdx.x;
    if (i >= ET) return;
    int s = i < E ? src[i] : i - E;
    int d = i < E ? dst[i] : i - E;
    float e = es[s] + ed[d];
    e = e > 0.0f ? e : LRELU_SLOPE * e;
    ebuf[i] = e;
    atomic_max_float(mbuf + d, e);
}

// pass2: ex = exp(e - m[dst]); segment sum into sbuf[dst]
__global__ void k_gat_e2(const int* __restrict__ src, const int* __restrict__ dst,
                         int E, int ET, const float* __restrict__ ebuf,
                         const float* __restrict__ mbuf, float* __restrict__ exbuf,
                         float* __restrict__ sbuf) {
    int i = blockIdx.x * blockDim.x + threadIdx.x;
    if (i >= ET) return;
    int d = i < E ? dst[i] : i - E;
    float ex = expf(ebuf[i] - mbuf[d]);
    exbuf[i] = ex;
    atomicAdd(sbuf + d, ex);
}

// pass3: out[dst] += (ex/s[dst]) * HW[src]   (wave per edge)
__global__ void k_gat_e3(const int* __restrict__ src, const int* __restrict__ dst,
                         int E, int ET, const float* __restrict__ exbuf,
                         const float* __restrict__ sbuf, const float* __restrict__ HW,
                         float* __restrict__ out, int F) {
    int wid  = (blockIdx.x * blockDim.x + threadIdx.x) >> 5;
    int lane = threadIdx.x & 31;
    if (wid >= ET) return;
    int s = wid < E ? src[wid] : wid - E;
    int d = wid < E ? dst[wid] : wid - E;
    float alpha = exbuf[wid] / sbuf[d];
    const float* hr = HW + (size_t)s * F;
    float* orow = out + (size_t)d * F;
    for (int f = lane; f < F; f += 32) atomicAdd(orow + f, alpha * hr[f]);
}

// ---------------- batch-norm (3-phase, coalesced) + ReLU ------------------
// phase 1: per-row-chunk partial sums. Wave reads 32 consecutive features per
// row iteration (coalesced); per-feature partials flushed with global atomics.
__global__ void k_bn_partial(const float* __restrict__ X, float* __restrict__ sums,
                             int R, int F) {
    int r0 = blockIdx.x * BN_ROWS;
    int rend = r0 + BN_ROWS;
    if (rend > R) rend = R;
    for (int f = threadIdx.x; f < F; f += TPB) {
        float s = 0.0f, q = 0.0f;
        for (int r = r0; r < rend; ++r) {
            float v = X[(size_t)r * F + f];
            s += v; q += v * v;
        }
        atomicAdd(sums + f, s);
        atomicAdd(sums + F + f, q);
    }
}

// phase 2: sums -> (mu, rstd) in place
__global__ void k_bn_stats(float* sums, int R, int F) {
    int f = blockIdx.x * blockDim.x + threadIdx.x;
    if (f < F) {
        float mu  = sums[f] / (float)R;
        float var = sums[F + f] / (float)R - mu * mu;   // biased var, as jnp.var
        sums[f]     = mu;
        sums[F + f] = rsqrtf(var + BN_EPS);
    }
}

// phase 3: elementwise normalize + ReLU, write fp32 + bf16  (F power of two)
__global__ void k_bn_apply(const float* __restrict__ X, float* __restrict__ H,
                           __bf16* __restrict__ Hb, const float* __restrict__ stats,
                           const float* __restrict__ gamma, const float* __restrict__ beta,
                           int R, int F) {
    int i = blockIdx.x * blockDim.x + threadIdx.x;
    if (i < R * F) {
        int f = i & (F - 1);
        float v = (X[i] - stats[f]) * stats[F + f] * gamma[f] + beta[f];
        v = fmaxf(v, 0.0f);
        H[i]  = v;
        Hb[i] = (__bf16)v;
    }
}

// ------------------------- global max pooling -----------------------------
__global__ void k_pool_max(const float* __restrict__ H, const int* __restrict__ batch,
                           float* __restrict__ pool, int R, int F) {
    int wid  = (blockIdx.x * blockDim.x + threadIdx.x) >> 5;
    int lane = threadIdx.x & 31;
    if (wid >= R) return;
    int g = batch[wid];
    const float* hr = H + (size_t)wid * F;
    float* pr = pool + (size_t)g * F;
    for (int f = lane; f < F; f += 32) atomic_max_float(pr + f, hr[f]);
}

// ------------------------- final FC (dout = 2) ----------------------------
__global__ void k_fc_out(const float* __restrict__ Gm, const float* __restrict__ W,
                         const float* __restrict__ b, float* __restrict__ out,
                         int K, int O) {
    __shared__ float red[TPB];
    int g = blockIdx.x / O;
    int o = blockIdx.x - g * O;
    float s = 0.0f;
    for (int k = threadIdx.x; k < K; k += TPB)
        s += Gm[(size_t)g * K + k] * W[(size_t)k * O + o];
    red[threadIdx.x] = s;
    __syncthreads();
    for (int w = TPB / 2; w > 0; w >>= 1) {
        if (threadIdx.x < w) red[threadIdx.x] += red[threadIdx.x + w];
        __syncthreads();
    }
    if (threadIdx.x == 0) out[g * O + o] = red[0] + b[o];
}

// ---------------------------------------------------------------------------
static const int DIMS[14] = {2, 64, 64, 128, 128, 256, 256, 256,
                             512, 512, 512, 512, 512, 512};

static inline int cdiv(long long a, long long b) { return (int)((a + b - 1) / b); }

extern "C" void kernel_launch(void* const* d_in, const int* in_sizes, int n_in,
                              void* d_out, int out_size, void* d_ws, size_t ws_size,
                              hipStream_t stream) {
    (void)n_in; (void)ws_size;
    const float* x    = (const float*)d_in[0];
    const int* ei     = (const int*)d_in[1];
    const int* batch  = (const int*)d_in[2];
    const int N  = in_sizes[0] / 2;
    const int E  = in_sizes[1] / 2;
    const int ET = E + N;
    const int G  = out_size / 2;
    const int* src = ei;
    const int* dst = ei + E;

    // input index map (setup_inputs dict order, lists flattened):
    // 3..15 Ws, 16..28 bs, 29..41 gammas, 42..54 betas,
    // 55..65 att_src, 66..76 att_dst, 77..79 fcW, 80..82 fcb
    const int FMAX = 512;

    // ---- workspace carve ----
    char* p = (char*)d_ws;
    auto carve = [&](size_t bytes) -> void* {
        void* r = (void*)p;
        p += (bytes + 255) & ~(size_t)255;
        return r;
    };
    float*  h    = (float*)carve((size_t)N * FMAX * 4);     // fp32 activations
    float*  hw   = (float*)carve((size_t)N * FMAX * 4);     // h @ W
    float*  agg  = (float*)carve((size_t)N * FMAX * 4);     // aggregated messages
    __bf16* hb   = (__bf16*)carve((size_t)N * FMAX * 2);    // bf16 copy of h
    __bf16* wt   = (__bf16*)carve((size_t)FMAX * FMAX * 2); // conv W^T bf16
    float*  dinv = (float*)carve((size_t)N * 4);
    float*  es   = (float*)carve((size_t)N * 4);
    float*  ed   = (float*)carve((size_t)N * 4);
    float*  mb   = (float*)carve((size_t)N * 4);
    float*  sb   = (float*)carve((size_t)N * 4);
    float*  ebuf = (float*)carve((size_t)ET * 4);
    float*  exbf = (float*)carve((size_t)ET * 4);
    float*  bnsum= (float*)carve((size_t)2 * FMAX * 4);     // BN partial sums
    // FC-stage aliases: hw & agg are dead after the conv stack + pooling reads h
    float*  pool = hw;                                      // G*512*4
    float*  g1   = (float*)((char*)hw + (1u << 20));        // G*4096*4 = 1MB
    float*  g2   = (float*)((char*)hw + (2u << 20));
    __bf16* gb   = (__bf16*)((char*)hw + (3u << 20));       // G*4096*2
    __bf16* wtfc = (__bf16*)agg;                            // up to 4096*4096*2 = 32MB

    // ---- GCN degree normalization: deg = 1 (self loop) + in-degree ----
    k_fill<<<cdiv(N, TPB), TPB, 0, stream>>>(dinv, 1.0f, N);
    k_deg<<<cdiv(E, TPB), TPB, 0, stream>>>(dst, E, dinv);
    k_rsqrt_inplace<<<cdiv(N, TPB), TPB, 0, stream>>>(dinv, N);

    // ---- 13 conv layers ----
    for (int i = 0; i < 13; ++i) {
        const int din  = DIMS[i];
        const int dout = DIMS[i + 1];
        const float* Wl  = (const float*)d_in[3 + i];
        const float* bl  = (const float*)d_in[16 + i];
        const float* gl  = (const float*)d_in[29 + i];
        const float* btl = (const float*)d_in[42 + i];

        // hw = h @ W
        if (i == 0) {
            k_gemm_k2<<<cdiv((long long)N * dout, TPB), TPB, 0, stream>>>(x, Wl, hw, N, dout);
        } else {
            k_wt_bf16<<<cdiv((long long)din * dout, TPB), TPB, 0, stream>>>(Wl, wt, din, dout);
            int grid = cdiv(N / 32, WPB) * (dout / 64);
            k_gemm_bf16<<<grid, TPB, 0, stream>>>(hb, wt, nullptr, hw, N, din, dout, 0);
        }

        // agg initialized with bias (segment-sum target)
        k_fill_bias<<<cdiv((long long)N * dout, TPB), TPB, 0, stream>>>(agg, bl, N, dout);

        if (i < 2) {
            // GCN: agg[dst] += norm * hw[src]
            k_gcn_agg<<<cdiv((long long)ET * 32, TPB), TPB, 0, stream>>>(
                src, dst, E, ET, dinv, hw, agg, dout);
        } else {
            // GAT: edge softmax over dst segments
            const float* as_ = (const float*)d_in[55 + (i - 2)];
            const float* ad_ = (const float*)d_in[66 + (i - 2)];
            k_gat_dots<<<cdiv((long long)N * 32, TPB), TPB, 0, stream>>>(hw, as_, ad_, es, ed, N, dout);
            k_fill<<<cdiv(N, TPB), TPB, 0, stream>>>(mb, -__builtin_inff(), N);
            k_fill<<<cdiv(N, TPB), TPB, 0, stream>>>(sb, 0.0f, N);
            k_gat_e1<<<cdiv(ET, TPB), TPB, 0, stream>>>(src, dst, E, ET, es, ed, ebuf, mb);
            k_gat_e2<<<cdiv(ET, TPB), TPB, 0, stream>>>(src, dst, E, ET, ebuf, mb, exbf, sb);
            k_gat_e3<<<cdiv((long long)ET * 32, TPB), TPB, 0, stream>>>(
                src, dst, E, ET, exbf, sb, hw, agg, dout);
        }

        // BN + ReLU -> h (fp32) and hb (bf16 for next GEMM)
        k_fill<<<cdiv(2 * dout, TPB), TPB, 0, stream>>>(bnsum, 0.0f, 2 * dout);
        k_bn_partial<<<cdiv(N, BN_ROWS), TPB, 0, stream>>>(agg, bnsum, N, dout);
        k_bn_stats<<<cdiv(dout, TPB), TPB, 0, stream>>>(bnsum, N, dout);
        k_bn_apply<<<cdiv((long long)N * dout, TPB), TPB, 0, stream>>>(
            agg, h, hb, bnsum, gl, btl, N, dout);
    }

    // ---- global max pool over batch segments -> pool[G,512] ----
    k_fill<<<cdiv(G * 512, TPB), TPB, 0, stream>>>(pool, -__builtin_inff(), G * 512);
    k_pool_max<<<cdiv((long long)N * 32, TPB), TPB, 0, stream>>>(h, batch, pool, N, 512);

    // ---- FC1: [G,512] @ [512,4096] + b, ReLU ----
    k_cvt_bf16<<<cdiv(G * 512, TPB), TPB, 0, stream>>>(pool, gb, G * 512);
    k_wt_bf16<<<cdiv(512 * 4096, TPB), TPB, 0, stream>>>((const float*)d_in[77], wtfc, 512, 4096);
    {
        int grid = cdiv(G / 32, WPB) * (4096 / 64);
        k_gemm_bf16<<<grid, TPB, 0, stream>>>(gb, wtfc, (const float*)d_in[80], g1, G, 512, 4096, 1);
    }

    // ---- FC2: [G,4096] @ [4096,4096] + b, ReLU ----
    k_cvt_bf16<<<cdiv(G * 4096, TPB), TPB, 0, stream>>>(g1, gb, G * 4096);
    k_wt_bf16<<<cdiv(4096 * 4096, TPB), TPB, 0, stream>>>((const float*)d_in[78], wtfc, 4096, 4096);
    {
        int grid = cdiv(G / 32, WPB) * (4096 / 64);
        k_gemm_bf16<<<grid, TPB, 0, stream>>>(gb, wtfc, (const float*)d_in[81], g2, G, 4096, 4096, 1);
    }

    // ---- FC3: [G,4096] @ [4096,2] + b -> d_out ----
    k_fc_out<<<G * 2, TPB, 0, stream>>>(g2, (const float*)d_in[79], (const float*)d_in[82],
                                        (float*)d_out, 4096, 2);
}
